// ALRDLinear_KIVI_Key_23596550324371
// MI455X (gfx1250) — matrix operand order repaired and checked by reading, not compiled, not run on hardware
//
#include <hip/hip_runtime.h>
#include <hip/hip_bf16.h>
#include <stdint.h>

typedef __attribute__((ext_vector_type(16))) __bf16          v16bf;
typedef __attribute__((ext_vector_type(2)))  __bf16          v2bf;
typedef __attribute__((ext_vector_type(2)))  float           v2f;
typedef __attribute__((ext_vector_type(8)))  float           v8f;
typedef __attribute__((ext_vector_type(16))) unsigned short  v16u;
typedef __attribute__((ext_vector_type(8)))  unsigned short  v8u;
typedef __attribute__((ext_vector_type(4)))  unsigned int    v4u;
typedef __attribute__((ext_vector_type(4)))  int             i32x4;

#define AS1 __attribute__((address_space(1)))
#define AS3 __attribute__((address_space(3)))

#if defined(__has_builtin)
#if __has_builtin(__builtin_amdgcn_global_load_async_to_lds_b128)
#define HAS_ASYNC_LDS 1
#endif
#endif
#ifndef HAS_ASYNC_LDS
#define HAS_ASYNC_LDS 0
#endif

// pack two fp32 -> packed bf16x2 (RNE): let the backend pick v_cvt_pk_bf16_f32
__device__ __forceinline__ unsigned pack2_bf16(float x, float y) {
    v2f f = {x, y};
    v2bf b = __builtin_convertvector(f, v2bf);
    return __builtin_bit_cast(unsigned, b);
}

__device__ __forceinline__ unsigned short f32_bf16(float f) {
    __bf16 b = (__bf16)f;  // RNE
    return __builtin_bit_cast(unsigned short, b);
}

// 16B global -> LDS copy: async DMA (ASYNCcnt) when available, sync fallback.
// Builtin prototype (probe-confirmed): (global int4*, local int4*, Ii offset, Ii cpol)
__device__ __forceinline__ void copy_b128(const void* g, void* l) {
#if HAS_ASYNC_LDS
    __builtin_amdgcn_global_load_async_to_lds_b128(
        (AS1 i32x4*)(void*)g, (AS3 i32x4*)l, 0, 0);
#else
    *(v4u*)l = *(const v4u*)g;
#endif
}

__device__ __forceinline__ void wait_async_lds() {
#if HAS_ASYNC_LDS
#if __has_builtin(__builtin_amdgcn_s_wait_asynccnt)
    __builtin_amdgcn_s_wait_asynccnt(0);
#else
    asm volatile("s_wait_asynccnt 0x0" ::: "memory");
#endif
#endif
}

// ---------------------------------------------------------------------------
// fp32 -> bf16 bulk convert (weights)
// ---------------------------------------------------------------------------
__global__ __launch_bounds__(256) void cvt_f32_bf16_kernel(
    const float* __restrict__ s, unsigned int* __restrict__ d, int n2) {
    int i = blockIdx.x * blockDim.x + threadIdx.x;
    int stride = gridDim.x * blockDim.x;
    for (; i < n2; i += stride) {
        float2 f = *(const float2*)(s + 2 * (size_t)i);
        d[i] = pack2_bf16(f.x, f.y);
    }
}

// ---------------------------------------------------------------------------
// Double-buffered bf16 WMMA GEMM:  C[M x NTOT] = A[M x KTOT] * Bw[NTOT x KTOT]^T (+bias)
//   CONV_A: A fp32, converted to bf16 during LDS staging (register path)
//   else  : A bf16, staged via async global->LDS DMA
// Workgroup tile 128x128, 8 wave32s, each wave 64x32 (4x2 WMMA frags), K-step 32.
// ---------------------------------------------------------------------------
template <int KTOT, int NTOT, bool CONV_A, bool BIAS>
__global__ __launch_bounds__(256) void gemm_bf16_kernel(
    const void* __restrict__ Aptr,
    const unsigned short* __restrict__ Bw,
    const float* __restrict__ bias,
    float* __restrict__ C) {
    constexpr int LDT = 40;        // padded LDS row stride (halfs)
    constexpr int T   = KTOT / 32; // K-steps
    __shared__ unsigned short lsA[2][128 * LDT];
    __shared__ unsigned short lsB[2][128 * LDT];

    const int tid  = threadIdx.x;
    const int lane = tid & 31;
    const int w    = tid >> 5;            // wave id (wave32)
    const int wm   = (w & 1) * 64;
    const int wn   = (w >> 1) * 32;
    const int m0   = blockIdx.y * 128;
    const int n0   = blockIdx.x * 128;
    const int lm   = lane & 15;
    const int lh   = lane >> 4;

    v8f acc[4][2];
#pragma unroll
    for (int mi = 0; mi < 4; ++mi)
#pragma unroll
        for (int ni = 0; ni < 2; ++ni) acc[mi][ni] = (v8f)0.f;

    // bf16 tile staging: thread -> row tid/2, two 16B chunks at col (tid&1)*16
    const int br = tid >> 1;
    const int bc = (tid & 1) << 4;

    auto stage_B = [&](int k0, int buf) {
        const unsigned short* g = Bw + (size_t)(n0 + br) * KTOT + k0 + bc;
        unsigned short* l = &lsB[buf][br * LDT + bc];
        copy_b128(g, l);
        copy_b128(g + 8, l + 8);
    };
    auto stage_A_bf16 = [&](int k0, int buf) {
        const unsigned short* A = (const unsigned short*)Aptr;
        const unsigned short* g = A + (size_t)(m0 + br) * KTOT + k0 + bc;
        unsigned short* l = &lsA[buf][br * LDT + bc];
        copy_b128(g, l);
        copy_b128(g + 8, l + 8);
    };
    auto load_A_f32 = [&](int k0, float4* pf) {
        const float* A = (const float*)Aptr;
#pragma unroll
        for (int i = 0; i < 4; ++i) {
            int j = tid + 256 * i;           // 1024 float4 slots, 128x32 fp32
            int r = j >> 3;
            int c = (j & 7) << 2;
            pf[i] = *(const float4*)(A + (size_t)(m0 + r) * KTOT + k0 + c);
        }
    };
    auto store_A_f32 = [&](const float4* pf, int buf) {
#pragma unroll
        for (int i = 0; i < 4; ++i) {
            int j = tid + 256 * i;
            int r = j >> 3;
            int c = (j & 7) << 2;
            unsigned p0 = pack2_bf16(pf[i].x, pf[i].y);
            unsigned p1 = pack2_bf16(pf[i].z, pf[i].w);
            *(uint2*)(&lsA[buf][r * LDT + c]) = make_uint2(p0, p1);
        }
    };

    // ---- prologue: stage tile 0 into buffer 0 ----
    if constexpr (CONV_A) {
        float4 pf[4];
        load_A_f32(0, pf);
        store_A_f32(pf, 0);
    } else {
        stage_A_bf16(0, 0);
    }
    stage_B(0, 0);
    wait_async_lds();
    __syncthreads();

    for (int t = 0; t < T; ++t) {
        const int buf  = t & 1;
        const int nbuf = buf ^ 1;
        const bool more = (t + 1) < T;
        float4 pf[4];
        if (more) {                       // issue next tile's loads first
            if constexpr (CONV_A) load_A_f32((t + 1) * 32, pf);
            else                  stage_A_bf16((t + 1) * 32, nbuf);
            stage_B((t + 1) * 32, nbuf);
        }

        // ---- fragments for tile t ----
        v16u af[4], bfr[2];
#pragma unroll
        for (int mi = 0; mi < 4; ++mi) {
            const unsigned short* p = &lsA[buf][(wm + mi * 16 + lm) * LDT + lh * 8];
            v8u lo = *(const v8u*)p;          // K = 8h .. 8h+7
            v8u hi = *(const v8u*)(p + 16);   // K = 16+8h .. 16+8h+7
            v16u x;
#pragma unroll
            for (int e = 0; e < 8; ++e) { x[e] = lo[e]; x[e + 8] = hi[e]; }
            af[mi] = x;
        }
#pragma unroll
        for (int ni = 0; ni < 2; ++ni) {
            const unsigned short* p = &lsB[buf][(wn + ni * 16 + lm) * LDT + lh * 8];
            v8u lo = *(const v8u*)p;          // K = 8h .. 8h+7
            v8u hi = *(const v8u*)(p + 16);   // K = 16+8h .. 16+8h+7
            v16u x;
#pragma unroll
            for (int e = 0; e < 8; ++e) { x[e] = lo[e]; x[e + 8] = hi[e]; }
            bfr[ni] = x;
        }

#pragma unroll
        for (int mi = 0; mi < 4; ++mi)
#pragma unroll
            for (int ni = 0; ni < 2; ++ni)
                acc[mi][ni] = __builtin_amdgcn_wmma_f32_16x16x32_bf16(
                    false, __builtin_bit_cast(v16bf, af[mi]),
                    false, __builtin_bit_cast(v16bf, bfr[ni]),
                    (short)0, acc[mi][ni], false, false);

        if constexpr (CONV_A) {
            if (more) store_A_f32(pf, nbuf);  // convert+store behind the WMMAs
        }
        wait_async_lds();
        __syncthreads();
    }

    // ---- epilogue: C/D layout VGPR v -> row v+8h, col l%16 ----
#pragma unroll
    for (int mi = 0; mi < 4; ++mi)
#pragma unroll
        for (int ni = 0; ni < 2; ++ni) {
            const int col = n0 + wn + ni * 16 + lm;
            if constexpr (BIAS) {
                const float b = bias[col];
#pragma unroll
                for (int v = 0; v < 8; ++v) {
                    const int row = m0 + wm + mi * 16 + lh * 8 + v;
                    C[(size_t)row * NTOT + col] = acc[mi][ni][v] + b;
                }
            } else {
#pragma unroll
                for (int v = 0; v < 8; ++v) {
                    const int row = m0 + wm + mi * 16 + lh * 8 + v;
                    C[(size_t)row * NTOT + col] = acc[mi][ni][v];
                }
            }
        }
}

// ---------------------------------------------------------------------------
// KIVI 2-bit per-channel quantize-dequantize along token dim (group 128),
// output bf16. One block = 128-row group x 64 columns. R = 512.
// ---------------------------------------------------------------------------
__global__ __launch_bounds__(256) void kivi_quant_kernel(
    const float* __restrict__ y, unsigned short* __restrict__ yq) {
    constexpr int R = 512;
    __shared__ float smn[4][64], smx[4][64], sscale[64], szero[64];

    const int c0 = blockIdx.x * 64;
    const int g  = blockIdx.y;
    const int c  = threadIdx.x & 63;
    const int q  = threadIdx.x >> 6;

    float vals[32];
    float mn = 3.4e38f, mx = -3.4e38f;
#pragma unroll
    for (int i = 0; i < 32; ++i) {
        float v = y[(size_t)(g * 128 + q * 32 + i) * R + c0 + c];
        vals[i] = v;
        mn = fminf(mn, v);
        mx = fmaxf(mx, v);
    }
    smn[q][c] = mn;
    smx[q][c] = mx;
    __syncthreads();
    if (q == 0) {
        float a = fminf(fminf(smn[0][c], smn[1][c]), fminf(smn[2][c], smn[3][c]));
        float b = fmaxf(fmaxf(smx[0][c], smx[1][c]), fmaxf(smx[2][c], smx[3][c]));
        sscale[c] = fmaxf((b - a) * (1.f / 3.f), 1e-8f);
        szero[c]  = a;
    }
    __syncthreads();
    const float scale = sscale[c];
    const float zp    = szero[c];
    const float inv   = 1.f / scale;
#pragma unroll
    for (int i = 0; i < 32; ++i) {
        float qv = rintf((vals[i] - zp) * inv);          // round half-even
        qv = fminf(fmaxf(qv, 0.f), 3.f);
        float dq = qv * scale + zp;
        yq[(size_t)(g * 128 + q * 32 + i) * R + c0 + c] = f32_bf16(dq);
    }
}

// ---------------------------------------------------------------------------
extern "C" void kernel_launch(void* const* d_in, const int* in_sizes, int n_in,
                              void* d_out, int out_size, void* d_ws,
                              size_t ws_size, hipStream_t stream) {
    const float* input = (const float*)d_in[0];   // [4,4096,4096]
    const float* W_B   = (const float*)d_in[1];   // [512,4096]
    const float* W_A   = (const float*)d_in[2];   // [4096,512]
    const float* b_A   = (const float*)d_in[3];   // [4096]
    float* out = (float*)d_out;                   // [4,4096,4096]

    constexpr int BS = 4 * 4096;   // flattened tokens (group 128 divides batch)
    constexpr int D  = 4096;
    constexpr int R  = 512;
    constexpr int O  = 4096;

    char* ws = (char*)d_ws;
    size_t off = 0;
    unsigned short* wb16 = (unsigned short*)(ws + off); off += (size_t)R * D * 2;
    unsigned short* wa16 = (unsigned short*)(ws + off); off += (size_t)O * R * 2;
    float*          ybuf = (float*)(ws + off);          off += (size_t)BS * R * 4;
    unsigned short* yq   = (unsigned short*)(ws + off); off += (size_t)BS * R * 2;
    if (ws_size < off) return;

    // 1) weights -> bf16 (L2-resident, reused by every tile)
    cvt_f32_bf16_kernel<<<2048, 256, 0, stream>>>(W_B, (unsigned int*)wb16, R * D / 2);
    cvt_f32_bf16_kernel<<<2048, 256, 0, stream>>>(W_A, (unsigned int*)wa16, O * R / 2);

    // 2) y = input @ W_B^T   (M=16384, K=4096, N=512)
    {
        dim3 grid(R / 128, BS / 128);
        gemm_bf16_kernel<D, R, true, false>
            <<<grid, 256, 0, stream>>>(input, wb16, nullptr, ybuf);
    }

    // 3) KIVI 2-bit quant-dequant along tokens, emit bf16
    {
        dim3 grid(R / 64, BS / 128);
        kivi_quant_kernel<<<grid, 256, 0, stream>>>(ybuf, yq);
    }

    // 4) out = yq @ W_A^T + b_A   (M=16384, K=512, N=4096)
    {
        dim3 grid(O / 128, BS / 128);
        gemm_bf16_kernel<R, O, false, true>
            <<<grid, 256, 0, stream>>>(yq, wa16, b_A, out);
    }
}
